// Attention_48893907697927
// MI455X (gfx1250) — matrix operand order repaired and checked by reading, not compiled, not run on hardware
//
#include <hip/hip_runtime.h>
#include <hip/hip_bf16.h>

typedef __attribute__((ext_vector_type(16))) _Float16 v16h;
typedef __attribute__((ext_vector_type(8)))  _Float16 v8h;
typedef __attribute__((ext_vector_type(8)))  float    v8f;

#define B_  4
#define QS_ 128
#define KS_ 1024
#define D_  1024
#define H_  256

// ---------------------------------------------------------------------------
// P[M,H] = X[M,D] * W[H,D]^T   (f32 in, f16 WMMA, f32 accumulate/out)
// STORE_T: also store per-batch transposed f16 copy PT[b][H][KS].
// One 16x16 output tile per wave; K-loop in steps of 32.
// ---------------------------------------------------------------------------
template <bool STORE_T>
__device__ __forceinline__ void proj_body(const float* __restrict__ X,
                                          const float* __restrict__ W,
                                          float* __restrict__ P,
                                          _Float16* __restrict__ PT, int M)
{
    const int lane = threadIdx.x & 31;
    const int wave = threadIdx.x >> 5;
    const int tile = blockIdx.x * 4 + wave;
    const int ntiles = (M / 16) * (H_ / 16);
    if (tile >= ntiles) return;

    const int mt = tile / (H_ / 16);
    const int nt = tile % (H_ / 16);
    const int m0 = mt * 16, n0 = nt * 16;
    const int hs = lane >> 4;      // half-wave select (0: lanes 0-15, 1: lanes 16-31)
    const int l  = lane & 15;

    const float* xrow = X + (size_t)(m0 + l) * D_;   // A row for this lane
    const float* wrow = W + (size_t)(n0 + l) * D_;   // B column == W row (W is [H,D])

    v8f c = {};
    for (int k0 = 0; k0 < D_; k0 += 32) {
        v16h af, bf;
        const int ka = k0 + hs * 8;
#pragma unroll
        for (int j = 0; j < 8; ++j) {
            af[j]     = (_Float16)xrow[ka + j];        // K = hs*8 + j
            af[8 + j] = (_Float16)xrow[ka + 16 + j];   // K = 16 + hs*8 + j
        }
        const int kb = k0 + hs * 8;
#pragma unroll
        for (int j = 0; j < 8; ++j) {
            bf[j]     = (_Float16)wrow[kb + j];        // K = hs*8 + j
            bf[8 + j] = (_Float16)wrow[kb + 16 + j];   // K = 16 + hs*8 + j
        }
        c = __builtin_amdgcn_wmma_f32_16x16x32_f16(false, af, false, bf,
                                                   (short)0, c, false, false);
    }

    const int n = n0 + l;
#pragma unroll
    for (int r = 0; r < 8; ++r) {
        const int m = m0 + r + hs * 8;
        P[(size_t)m * H_ + n] = c[r];
        if constexpr (STORE_T) {
            const int bb = m / KS_, kin = m % KS_;     // KS_ multiple of 16 -> tile in one batch
            PT[((size_t)bb * H_ + n) * KS_ + kin] = (_Float16)c[r];
        }
    }
}

__global__ void proj_kernel(const float* __restrict__ X, const float* __restrict__ W,
                            float* __restrict__ P, int M)
{
    proj_body<false>(X, W, P, nullptr, M);
}

__global__ void proj_kernel_t(const float* __restrict__ X, const float* __restrict__ W,
                              float* __restrict__ P, _Float16* __restrict__ PT, int M)
{
    proj_body<true>(X, W, P, PT, M);
}

// ---------------------------------------------------------------------------
// One workgroup (256 thr = 8 waves) per (b,q):
// scores[k] = sum_h tanh(Qp[q,h] + Kp[k,h] + bias[h]) * v[h],  v = a/||a|| * scal
// then softmax over k. Each thread owns 4 k-rows; Kp streamed from L2.
// ---------------------------------------------------------------------------
__global__ void scores_softmax_kernel(const float* __restrict__ Qp, const float* __restrict__ Kp,
                                      const float* __restrict__ att, const float* __restrict__ scal,
                                      const float* __restrict__ bias,
                                      float* __restrict__ probs, _Float16* __restrict__ probs16)
{
    __shared__ float qsh[H_];
    __shared__ float vsh[H_];
    __shared__ float red[H_];

    const int t = threadIdx.x;           // 0..255
    const int b = blockIdx.x / QS_;
    const int q = blockIdx.x % QS_;

    qsh[t] = Qp[((size_t)b * QS_ + q) * H_ + t] + bias[t];
    const float av = att[t];
    red[t] = av * av;
    __syncthreads();
    for (int s = 128; s > 0; s >>= 1) { if (t < s) red[t] += red[t + s]; __syncthreads(); }
    const float vscale = scal[0] * __frsqrt_rn(red[0]);
    __syncthreads();
    vsh[t] = av * vscale;
    __syncthreads();

    const float* kr0 = Kp + ((size_t)b * KS_ + t) * H_;
    const float* kr1 = kr0 + 256 * H_;
    const float* kr2 = kr0 + 512 * H_;
    const float* kr3 = kr0 + 768 * H_;

    float s0 = 0.f, s1 = 0.f, s2 = 0.f, s3 = 0.f;
    for (int h = 0; h < H_; h += 4) {
        const float4 k0v = *(const float4*)(kr0 + h);
        const float4 k1v = *(const float4*)(kr1 + h);
        const float4 k2v = *(const float4*)(kr2 + h);
        const float4 k3v = *(const float4*)(kr3 + h);
#pragma unroll
        for (int j = 0; j < 4; ++j) {
            const float qb = qsh[h + j];
            const float vv = vsh[h + j];
            const float* k0p = &k0v.x; const float* k1p = &k1v.x;
            const float* k2p = &k2v.x; const float* k3p = &k3v.x;
            s0 += tanhf(qb + k0p[j]) * vv;
            s1 += tanhf(qb + k1p[j]) * vv;
            s2 += tanhf(qb + k2p[j]) * vv;
            s3 += tanhf(qb + k3p[j]) * vv;
        }
    }

    // softmax over the 1024 scores held 4-per-thread
    float m = fmaxf(fmaxf(s0, s1), fmaxf(s2, s3));
    red[t] = m; __syncthreads();
    for (int s = 128; s > 0; s >>= 1) { if (t < s) red[t] = fmaxf(red[t], red[t + s]); __syncthreads(); }
    const float mx = red[0];
    __syncthreads();
    const float e0 = expf(s0 - mx), e1 = expf(s1 - mx);
    const float e2 = expf(s2 - mx), e3 = expf(s3 - mx);
    red[t] = e0 + e1 + e2 + e3; __syncthreads();
    for (int s = 128; s > 0; s >>= 1) { if (t < s) red[t] += red[t + s]; __syncthreads(); }
    const float inv = 1.0f / red[0];

    const size_t base = ((size_t)b * QS_ + q) * KS_;
    const float p0 = e0 * inv, p1 = e1 * inv, p2 = e2 * inv, p3 = e3 * inv;
    probs[base + t]        = p0;  probs16[base + t]        = (_Float16)p0;
    probs[base + t + 256]  = p1;  probs16[base + t + 256]  = (_Float16)p1;
    probs[base + t + 512]  = p2;  probs16[base + t + 512]  = (_Float16)p2;
    probs[base + t + 768]  = p3;  probs16[base + t + 768]  = (_Float16)p3;
}

// ---------------------------------------------------------------------------
// context[b,q,h] = sum_k probs[b,q,k] * Kp[b,k,h]
// A = probs16 [b][QS][KS] row-major, B via KpT16 [b][H][KS] (K-contiguous).
// ---------------------------------------------------------------------------
__global__ void context_kernel(const _Float16* __restrict__ probs16,
                               const _Float16* __restrict__ KpT,
                               float* __restrict__ ctx)
{
    const int lane = threadIdx.x & 31;
    const int wave = threadIdx.x >> 5;
    const int tile = blockIdx.x * 4 + wave;              // 512 tiles total
    if (tile >= B_ * (QS_ / 16) * (H_ / 16)) return;

    const int per_batch = (QS_ / 16) * (H_ / 16);        // 128
    const int b   = tile / per_batch;
    const int rem = tile % per_batch;
    const int m0 = (rem / (H_ / 16)) * 16;
    const int n0 = (rem % (H_ / 16)) * 16;
    const int hs = lane >> 4;
    const int l  = lane & 15;

    const _Float16* arow = probs16 + ((size_t)b * QS_ + m0 + l) * KS_;
    const _Float16* brow = KpT     + ((size_t)b * H_  + n0 + l) * KS_;

    v8f c = {};
    for (int k0 = 0; k0 < KS_; k0 += 32) {
        const v8h alo = *(const v8h*)(arow + k0 + hs * 8);
        const v8h ahi = *(const v8h*)(arow + k0 + 16 + hs * 8);
        const v8h blo = *(const v8h*)(brow + k0 + hs * 8);
        const v8h bhi = *(const v8h*)(brow + k0 + 16 + hs * 8);
        v16h af, bf;
#pragma unroll
        for (int j = 0; j < 8; ++j) {
            af[j] = alo[j]; af[8 + j] = ahi[j];
            bf[j] = blo[j]; bf[8 + j] = bhi[j];
        }
        c = __builtin_amdgcn_wmma_f32_16x16x32_f16(false, af, false, bf,
                                                   (short)0, c, false, false);
    }

#pragma unroll
    for (int r = 0; r < 8; ++r)
        ctx[((size_t)b * QS_ + m0 + r + hs * 8) * H_ + (n0 + l)] = c[r];
}

// ---------------------------------------------------------------------------
extern "C" void kernel_launch(void* const* d_in, const int* in_sizes, int n_in,
                              void* d_out, int out_size, void* d_ws, size_t ws_size,
                              hipStream_t stream) {
    (void)in_sizes; (void)n_in; (void)out_size; (void)ws_size;

    const float* Q    = (const float*)d_in[0];
    const float* K    = (const float*)d_in[1];
    const float* Wq   = (const float*)d_in[2];
    const float* Wk   = (const float*)d_in[3];
    const float* att  = (const float*)d_in[4];
    const float* scal = (const float*)d_in[5];
    const float* bias = (const float*)d_in[6];

    // workspace layout
    float*    Qp  = (float*)d_ws;                              // 512*256 f32
    float*    Kp  = Qp + (size_t)B_ * QS_ * H_;                // 4096*256 f32
    _Float16* KpT = (_Float16*)(Kp + (size_t)B_ * KS_ * H_);   // [b][H][KS] f16
    _Float16* p16 = KpT + (size_t)B_ * H_ * KS_;               // [b][QS][KS] f16

    float* ctx   = (float*)d_out;                              // B*QS*H
    float* probs = ctx + (size_t)B_ * QS_ * H_;                // B*QS*KS

    // Qp: 512 tiles -> 128 blocks of 4 waves
    proj_kernel<<<dim3(128), dim3(128), 0, stream>>>(Q, Wq, Qp, B_ * QS_);
    // Kp (+ transposed f16 copy): 4096 tiles -> 1024 blocks
    proj_kernel_t<<<dim3(1024), dim3(128), 0, stream>>>(K, Wk, Kp, KpT, B_ * KS_);
    // fused tanh-scores + softmax: one WG per (b,q)
    scores_softmax_kernel<<<dim3(B_ * QS_), dim3(H_), 0, stream>>>(Qp, Kp, att, scal, bias,
                                                                   probs, p16);
    // context GEMM: 512 tiles -> 128 blocks
    context_kernel<<<dim3(128), dim3(128), 0, stream>>>(p16, KpT, ctx);
}
